// Attend_72086731096112
// MI455X (gfx1250) — compile-verified, matrix operand order checked by reading, not run on hardware
//
#include <hip/hip_runtime.h>

#define BATCH   2
#define SEQ     2048
#define DMODEL  1024
#define NHEADS  16
#define HD      64
#define MROWS   (BATCH * SEQ)

typedef __attribute__((ext_vector_type(16))) __bf16 v16bf;
typedef __attribute__((ext_vector_type(8)))  float  v8f;

// float -> bf16 bits, round-to-nearest-even
static __device__ __forceinline__ unsigned short f2bf(float f) {
  unsigned int u = __builtin_bit_cast(unsigned int, f);
  unsigned int r = u + 0x7FFFu + ((u >> 16) & 1u);
  return (unsigned short)(r >> 16);
}

// 16x32 bf16 WMMA operand fragment = two contiguous 16B chunks:
// lane supplies one row/col; K-pairs for VGPRs 0..3 live at [p, p+8),
// VGPRs 4..7 at [p+16, p+24)  (ISA 7.12.2 A-layout, h folded into p).
struct U128 { unsigned int x, y, z, w; };
struct U256 { U128 lo, hi; };
static __device__ __forceinline__ v16bf ldfrag(const unsigned short* p) {
  U256 u;
  u.lo = *(const U128*)(p);
  u.hi = *(const U128*)(p + 16);
  return __builtin_bit_cast(v16bf, u);
}

#define WMMA_BF16(A, B, C)                                                   \
  __builtin_amdgcn_wmma_f32_16x16x32_bf16(false, (A), false, (B), (short)0,  \
                                          (C), false, false)

// Max-reduce over the 16-lane half via DPP (pure VALU, no LDS pipe).
// Masks {1,2,7,15}: quad_perm(1,0,3,2)=0xB1, quad_perm(2,3,0,1)=0x4E,
// row_half_mirror=0x141 (xor7), row_mirror=0x140 (xor15) generate the
// full 16-lane group.
static __device__ __forceinline__ float dpp_max16(float r) {
  int t;
  t = __builtin_amdgcn_mov_dpp(__builtin_bit_cast(int, r), 0xB1, 0xF, 0xF, true);
  r = fmaxf(r, __builtin_bit_cast(float, t));
  t = __builtin_amdgcn_mov_dpp(__builtin_bit_cast(int, r), 0x4E, 0xF, 0xF, true);
  r = fmaxf(r, __builtin_bit_cast(float, t));
  t = __builtin_amdgcn_mov_dpp(__builtin_bit_cast(int, r), 0x141, 0xF, 0xF, true);
  r = fmaxf(r, __builtin_bit_cast(float, t));
  t = __builtin_amdgcn_mov_dpp(__builtin_bit_cast(int, r), 0x140, 0xF, 0xF, true);
  r = fmaxf(r, __builtin_bit_cast(float, t));
  return r;
}

// ---------------------------------------------------------------------------
// x (f32, [M,K]) -> bf16 row-major, 4 elems/thread
// ---------------------------------------------------------------------------
__global__ __launch_bounds__(256) void xconv_kernel(
    const float* __restrict__ x, unsigned short* __restrict__ o) {
  size_t i = ((size_t)blockIdx.x * 256 + threadIdx.x) * 4;
  float a = x[i], b = x[i + 1], c = x[i + 2], d = x[i + 3];
  unsigned int lo = (unsigned int)f2bf(a) | ((unsigned int)f2bf(b) << 16);
  unsigned int hi = (unsigned int)f2bf(c) | ((unsigned int)f2bf(d) << 16);
  *(unsigned int*)(o + i)     = lo;
  *(unsigned int*)(o + i + 2) = hi;
}

// ---------------------------------------------------------------------------
// W (f32, [K,N]) -> bf16 transposed [N,K], LDS-tiled 32x32 so both sides
// stay coalesced. Grid (N/32, K/32), 256 threads.
// ---------------------------------------------------------------------------
__global__ __launch_bounds__(256) void wconv_kernel(
    const float* __restrict__ W, unsigned short* __restrict__ Wt) {
  __shared__ float t[32][33];
  const int tx = threadIdx.x & 31;
  const int ty = threadIdx.x >> 5;  // 0..7
  const int n0 = blockIdx.x * 32;
  const int k0 = blockIdx.y * 32;
#pragma unroll
  for (int r = 0; r < 32; r += 8)
    t[ty + r][tx] = W[(size_t)(k0 + ty + r) * DMODEL + n0 + tx];
  __syncthreads();
#pragma unroll
  for (int r = 0; r < 32; r += 8)
    Wt[(size_t)(n0 + ty + r) * DMODEL + k0 + tx] = f2bf(t[tx][ty + r]);
}

// ---------------------------------------------------------------------------
// GEMM: C[M,N] = Abf[M,K] @ WtBf[N,K]^T + bias, optional scale.
// Wave computes a 64x64 tile: 4 A-frags x 4 B-frags -> 16 WMMAs per 16
// global_load_b128 per k-step (1.0 WMMA/load, 32 FLOP/byte from L2).
// No LDS: the 192MB L2 holds the whole working set and provides reuse.
// MODE 0: f32 out [M,N]       (final projection -> d_out)
// MODE 1: bf16 out [B,H,S,hd] (Q, K)
// MODE 2: bf16 out [B,H,hd,S] (V transposed, for contiguous P@V fragments)
// ---------------------------------------------------------------------------
template <int MODE>
__global__ __launch_bounds__(128) void gemm_kernel(
    const unsigned short* __restrict__ Abf,
    const unsigned short* __restrict__ Wt,
    const float* __restrict__ bias, float scale,
    float* __restrict__ fout, unsigned short* __restrict__ uout) {
  const int lane = threadIdx.x & 31;
  const int wave = threadIdx.x >> 5;  // 0..3
  const int n = lane & 15;
  const int h = lane >> 4;
  const int hoff = h * 8;
  const int rbase = blockIdx.x * 128 + (wave >> 1) * 64;
  const int cbase = blockIdx.y * 128 + (wave & 1) * 64;

  const unsigned short* arow[4];
  const unsigned short* brow[4];
#pragma unroll
  for (int t = 0; t < 4; ++t) {
    arow[t] = Abf + (size_t)(rbase + t * 16 + n) * DMODEL + hoff;
    brow[t] = Wt + (size_t)(cbase + t * 16 + n) * DMODEL + hoff;
  }

  v8f acc[4][4] = {};
  for (int k0 = 0; k0 < DMODEL; k0 += 32) {
    v16bf af[4], bf[4];
#pragma unroll
    for (int t = 0; t < 4; ++t) af[t] = ldfrag(arow[t] + k0);
#pragma unroll
    for (int t = 0; t < 4; ++t) bf[t] = ldfrag(brow[t] + k0);
#pragma unroll
    for (int i = 0; i < 4; ++i)
#pragma unroll
      for (int j = 0; j < 4; ++j)
        acc[i][j] = WMMA_BF16(af[i], bf[j], acc[i][j]);
  }

#pragma unroll
  for (int rt = 0; rt < 4; ++rt) {
#pragma unroll
    for (int ct = 0; ct < 4; ++ct) {
      const int col = cbase + ct * 16 + n;
      const float bv = bias[col];
      const int head = col >> 6;  // / HD
      const int c = col & 63;
#pragma unroll
      for (int v = 0; v < 8; ++v) {
        int row = rbase + rt * 16 + v + 8 * h;
        float val = (acc[rt][ct][v] + bv) * scale;
        if (MODE == 0) {
          fout[(size_t)row * DMODEL + col] = val;
        } else {
          int bb = row >> 11;  // / SEQ
          int s = row & 2047;  // % SEQ
          if (MODE == 1)
            uout[(((size_t)bb * NHEADS + head) * SEQ + s) * HD + c] = f2bf(val);
          else
            uout[(((size_t)bb * NHEADS + head) * HD + c) * SEQ + s] = f2bf(val);
        }
      }
    }
  }
}

// ---------------------------------------------------------------------------
// One 32-key flash block for a 32-query wave tile. MASK=true only for the
// single diagonal block (kb0 == q0); all earlier blocks are mask-free.
// Row max via DPP (VALU); row sum via WMMA against an all-ones fragment.
// ---------------------------------------------------------------------------
template <bool MASK>
static __device__ __forceinline__ void flash_step(
    int kb0, int q0, int n, int h, int hoff,
    const unsigned short* __restrict__ Kb,
    const unsigned short* __restrict__ Vb, unsigned short* pl,
    const v16bf (&qa)[2][2], const v16bf& onesf, float (&mrow)[2][8],
    float (&lrow)[2][8], v8f (&oacc)[2][4]) {
  // ---- S = Q K^T over 32 keys: 8 WMMAs ----
  v16bf kf[2][2];
#pragma unroll
  for (int ks = 0; ks < 2; ++ks) {
    const unsigned short* krow = Kb + (size_t)(kb0 + ks * 16 + n) * HD + hoff;
    kf[ks][0] = ldfrag(krow);
    kf[ks][1] = ldfrag(krow + 32);
  }
  v8f s[2][2] = {};
#pragma unroll
  for (int qs = 0; qs < 2; ++qs)
#pragma unroll
    for (int ks = 0; ks < 2; ++ks) {
      s[qs][ks] = WMMA_BF16(qa[qs][0], kf[ks][0], s[qs][ks]);
      s[qs][ks] = WMMA_BF16(qa[qs][1], kf[ks][1], s[qs][ks]);
    }

#pragma unroll
  for (int qs = 0; qs < 2; ++qs) {
    // ---- (mask +) per-row max via DPP butterfly in the 16-lane half ----
    float rmax[8];
#pragma unroll
    for (int v = 0; v < 8; ++v) {
      if (MASK) {
        int qrow_i = q0 + qs * 16 + v + 8 * h;
        if (kb0 + n > qrow_i)      s[qs][0][v] = -3.0e38f;
        if (kb0 + 16 + n > qrow_i) s[qs][1][v] = -3.0e38f;
      }
      rmax[v] = dpp_max16(fmaxf(s[qs][0][v], s[qs][1][v]));
    }
    // ---- online softmax: rescale O/l, compute P, stage P to LDS ----
#pragma unroll
    for (int v = 0; v < 8; ++v) {
      float mn = fmaxf(mrow[qs][v], rmax[v]);
      float sc = __expf(mrow[qs][v] - mn);
      mrow[qs][v] = mn;
      lrow[qs][v] *= sc;
      oacc[qs][0][v] *= sc;
      oacc[qs][1][v] *= sc;
      oacc[qs][2][v] *= sc;
      oacc[qs][3][v] *= sc;
      float p0 = __expf(s[qs][0][v] - mn);
      float p1 = __expf(s[qs][1][v] - mn);
      int m = qs * 16 + v + 8 * h;
      pl[m * 32 + n]      = f2bf(p0);
      pl[m * 32 + 16 + n] = f2bf(p1);
    }
  }

  __builtin_amdgcn_wave_barrier();
  asm volatile("s_wait_dscnt 0" ::: "memory");

  // ---- P back in A-operand layout (2 x 16 rows x 32 keys) ----
  v16bf pa0 = ldfrag(pl + (size_t)n * 32 + hoff);
  v16bf pa1 = ldfrag(pl + (size_t)(16 + n) * 32 + hoff);

  // ---- row sums: P x ones via WMMA (lands per-row in C layout) ----
  v8f zero = {};
  v8f ls0 = WMMA_BF16(pa0, onesf, zero);
  v8f ls1 = WMMA_BF16(pa1, onesf, zero);
#pragma unroll
  for (int v = 0; v < 8; ++v) {
    lrow[0][v] += ls0[v];
    lrow[1][v] += ls1[v];
  }

  // ---- O += P @ V : V^T fragments reused by both q-subtiles, 8 WMMAs ----
#pragma unroll
  for (int dc = 0; dc < 4; ++dc) {
    v16bf vb = ldfrag(Vb + (size_t)(dc * 16 + n) * SEQ + kb0 + hoff);
    oacc[0][dc] = WMMA_BF16(pa0, vb, oacc[0][dc]);
    oacc[1][dc] = WMMA_BF16(pa1, vb, oacc[1][dc]);
  }
  __builtin_amdgcn_wave_barrier();  // LDS reuse next block
}

// ---------------------------------------------------------------------------
// Flash attention: one wave = 32 queries of one (b,h). Mask-free main loop
// over full key blocks, single masked diagonal block as tail.
// Q,K bf16 [B,H,S,hd]; V bf16 transposed [B,H,hd,S]; out bf16 [B,S,D].
// ---------------------------------------------------------------------------
__global__ __launch_bounds__(256) void flash_attn_kernel(
    const unsigned short* __restrict__ Q, const unsigned short* __restrict__ K,
    const unsigned short* __restrict__ Vt,
    unsigned short* __restrict__ O) {
  __shared__ alignas(32) unsigned short sP[8][32 * 32];  // per-wave P staging
  const int lane = threadIdx.x & 31;
  const int wave = threadIdx.x >> 5;
  const int n = lane & 15;
  const int h = lane >> 4;
  const int hoff = h * 8;
  const int gw = blockIdx.x * 8 + wave;
  const int qt = gw & 63;  // S/32 query tiles
  const int bh = gw >> 6;  // B*H
  const int q0 = qt * 32;

  const unsigned short* Qb = Q + (size_t)bh * SEQ * HD;
  const unsigned short* Kb = K + (size_t)bh * SEQ * HD;
  const unsigned short* Vb = Vt + (size_t)bh * HD * SEQ;
  unsigned short* pl = sP[wave];

  // Q as A-operand: 2 q-subtiles x 2 hd-chunks
  v16bf qa[2][2];
#pragma unroll
  for (int qs = 0; qs < 2; ++qs) {
    const unsigned short* qrow = Qb + (size_t)(q0 + qs * 16 + n) * HD + hoff;
    qa[qs][0] = ldfrag(qrow);
    qa[qs][1] = ldfrag(qrow + 32);
  }
  // all-ones bf16 B fragment for row sums
  v16bf onesf;
#pragma unroll
  for (int t = 0; t < 16; ++t)
    onesf[t] = __builtin_bit_cast(__bf16, (unsigned short)0x3F80);

  float mrow[2][8], lrow[2][8];
#pragma unroll
  for (int qs = 0; qs < 2; ++qs)
#pragma unroll
    for (int v = 0; v < 8; ++v) { mrow[qs][v] = -3.0e38f; lrow[qs][v] = 0.0f; }
  v8f oacc[2][4] = {};

  // Mask-free main loop: every key of block kb0 < q0 precedes every query.
  for (int kb0 = 0; kb0 < q0; kb0 += 32)
    flash_step<false>(kb0, q0, n, h, hoff, Kb, Vb, pl, qa, onesf, mrow, lrow,
                      oacc);
  // Single diagonal block.
  flash_step<true>(q0, q0, n, h, hoff, Kb, Vb, pl, qa, onesf, mrow, lrow, oacc);

  // ---- normalize and store bf16 [B, S, head*HD + d] ----
  const int bb = bh >> 4;    // / NHEADS
  const int head = bh & 15;  // % NHEADS
#pragma unroll
  for (int qs = 0; qs < 2; ++qs)
#pragma unroll
    for (int v = 0; v < 8; ++v) {
      float inv = 1.0f / lrow[qs][v];
      int srow = q0 + qs * 16 + v + 8 * h;
      size_t base = ((size_t)bb * SEQ + srow) * DMODEL + head * HD;
      O[base + 0  + n] = f2bf(oacc[qs][0][v] * inv);
      O[base + 16 + n] = f2bf(oacc[qs][1][v] * inv);
      O[base + 32 + n] = f2bf(oacc[qs][2][v] * inv);
      O[base + 48 + n] = f2bf(oacc[qs][3][v] * inv);
    }
}

// ---------------------------------------------------------------------------
extern "C" void kernel_launch(void* const* d_in, const int* in_sizes, int n_in,
                              void* d_out, int out_size, void* d_ws,
                              size_t ws_size, hipStream_t stream) {
  (void)in_sizes; (void)n_in; (void)out_size; (void)ws_size;
  const float* x  = (const float*)d_in[0];
  const float* Wq = (const float*)d_in[1];
  const float* bq = (const float*)d_in[2];
  const float* Wk = (const float*)d_in[3];
  const float* bk = (const float*)d_in[4];
  const float* Wv = (const float*)d_in[5];
  const float* bv = (const float*)d_in[6];
  const float* Wo = (const float*)d_in[7];
  const float* bo = (const float*)d_in[8];

  const size_t XE = (size_t)MROWS * DMODEL;    // 4M elems
  const size_t WE = (size_t)DMODEL * DMODEL;   // 1M elems
  unsigned short* xb  = (unsigned short*)d_ws;  // bf16 x
  unsigned short* WqT = xb + XE;
  unsigned short* WkT = WqT + WE;
  unsigned short* WvT = WkT + WE;
  unsigned short* WoT = WvT + WE;
  unsigned short* Qw  = WoT + WE;   // [B,H,S,hd]
  unsigned short* Kw  = Qw + XE;    // [B,H,S,hd]
  unsigned short* Vtw = Kw + XE;    // [B,H,hd,S]
  unsigned short* Aw  = Vtw + XE;   // [B,S,D]
  // total: 5*XE + 4*WE = 24M elems = 48 MB of workspace

  // One-time conversions (keeps all f2bf out of the GEMM hot loops)
  xconv_kernel<<<(int)(XE / (256 * 4)), 256, 0, stream>>>(x, xb);
  dim3 wt(DMODEL / 32, DMODEL / 32);
  wconv_kernel<<<wt, 256, 0, stream>>>(Wq, WqT);
  wconv_kernel<<<wt, 256, 0, stream>>>(Wk, WkT);
  wconv_kernel<<<wt, 256, 0, stream>>>(Wv, WvT);
  wconv_kernel<<<wt, 256, 0, stream>>>(Wo, WoT);

  dim3 gg(MROWS / 128, DMODEL / 128);  // 32 x 8 blocks, 4 waves each
  gemm_kernel<1><<<gg, 128, 0, stream>>>(xb, WqT, bq, 0.125f, nullptr, Qw);
  gemm_kernel<1><<<gg, 128, 0, stream>>>(xb, WkT, bk, 1.0f, nullptr, Kw);
  gemm_kernel<2><<<gg, 128, 0, stream>>>(xb, WvT, bv, 1.0f, nullptr, Vtw);

  const int totalWaves = BATCH * NHEADS * (SEQ / 32);  // 2048
  flash_attn_kernel<<<totalWaves / 8, 256, 0, stream>>>(Qw, Kw, Vtw, Aw);

  gemm_kernel<0><<<gg, 128, 0, stream>>>(Aw, WoT, bo, 1.0f, (float*)d_out, nullptr);
}
